// DividedSpaceTimeBlock_41566693491461
// MI455X (gfx1250) — matrix operand order repaired and checked by reading, not compiled, not run on hardware
//
#include <hip/hip_runtime.h>
#include <math.h>

// ---------------- problem constants (fixed by reference) ----------------
#define DIMC    768
#define HEADS_C 12
#define HIDDENC 3072
#define BC      16
#define FC      8
#define NSPC    196
#define NTOK    1569          // 1 + FC*NSPC
#define HD      64            // head dim = DIMC/HEADS_C
#define QKVC    2304          // 3*DIMC

// GEMM tiling
#define MT 64                 // block tile M
#define NT 128                // block tile N
#define KT 32                 // K step (one WMMA K)
#define LDS_STRIDE 40         // bf16 elems per LDS row (32 data + 8 pad = 80B)

typedef __bf16 bf16_t;
typedef bf16_t v16bf __attribute__((ext_vector_type(16)));
typedef bf16_t v8bf  __attribute__((ext_vector_type(8)));
typedef float  v8f   __attribute__((ext_vector_type(8)));
typedef int    vsi4  __attribute__((vector_size(16)));   // async-LDS builtin pointee type

#if defined(__has_builtin)
#if __has_builtin(__builtin_amdgcn_global_load_async_to_lds_b128) && \
    __has_builtin(__builtin_amdgcn_s_wait_asynccnt)
#define HAVE_ASYNC_LDS 1
#endif
#endif

static __device__ __forceinline__ void copy_b128_to_lds(const bf16_t* g, bf16_t* l) {
#ifdef HAVE_ASYNC_LDS
    // param0: v4i in AS(1) (global); param1: v4i in AS(3) (LDS, 32-bit offset)
    __builtin_amdgcn_global_load_async_to_lds_b128(
        (__attribute__((address_space(1))) vsi4*)(uintptr_t)g,
        (__attribute__((address_space(3))) vsi4*)(uint32_t)(uintptr_t)l,
        0, 0);
#else
    *(v8bf*)l = *(const v8bf*)g;
#endif
}

static __device__ __forceinline__ void wait_async_copies() {
#ifdef HAVE_ASYNC_LDS
    __builtin_amdgcn_s_wait_asynccnt(0);
#endif
}

static __device__ __forceinline__ bf16_t f2bf(float f) {
    unsigned u = __builtin_bit_cast(unsigned, f);
    unsigned r = u + 0x7FFFu + ((u >> 16) & 1u);     // round-to-nearest-even
    unsigned short h = (unsigned short)(r >> 16);
    return __builtin_bit_cast(bf16_t, h);
}

static __device__ __forceinline__ float gelu_exact(float x) {
    return 0.5f * x * (1.0f + erff(x * 0.70710678118654752f));
}

// ---------------- fp32 -> bf16 elementwise ----------------
__global__ void cvt_f32_bf16_kernel(const float* __restrict__ src,
                                    bf16_t* __restrict__ dst, size_t n) {
    size_t i = (size_t)blockIdx.x * blockDim.x + threadIdx.x;
    size_t stride = (size_t)gridDim.x * blockDim.x;
    for (; i < n; i += stride) dst[i] = f2bf(src[i]);
}

// ---------------- layernorm over DIMC, bf16 output ----------------
__global__ void layernorm_bf16_kernel(const float* __restrict__ x,
                                      const float* __restrict__ g,
                                      const float* __restrict__ be,
                                      bf16_t* __restrict__ out) {
    __shared__ float red[256];
    const int row = blockIdx.x;
    const int tid = threadIdx.x;
    const float* xr = x + (size_t)row * DIMC;

    float s = 0.f;
    for (int i = tid; i < DIMC; i += 256) s += xr[i];
    red[tid] = s; __syncthreads();
    for (int o = 128; o > 0; o >>= 1) { if (tid < o) red[tid] += red[tid + o]; __syncthreads(); }
    const float mean = red[0] * (1.0f / DIMC);
    __syncthreads();

    float v = 0.f;
    for (int i = tid; i < DIMC; i += 256) { float d = xr[i] - mean; v += d * d; }
    red[tid] = v; __syncthreads();
    for (int o = 128; o > 0; o >>= 1) { if (tid < o) red[tid] += red[tid + o]; __syncthreads(); }
    const float rstd = rsqrtf(red[0] * (1.0f / DIMC) + 1e-5f);

    bf16_t* orow = out + (size_t)row * DIMC;
    for (int i = tid; i < DIMC; i += 256)
        orow[i] = f2bf((xr[i] - mean) * rstd * g[i] + be[i]);
}

// ---------------- LDS-tiled bf16 WMMA GEMM:  out[M,Nc] = A[M,K] * W[Nc,K]^T ----------------
// Block: 256 threads = 8 waves as 2(M)x4(N); block tile 64x128, wave tile 32x32
// (2x2 v8f accumulators). A/B K-slabs (64x32, 128x32 bf16) staged in LDS via
// GLOBAL_LOAD_ASYNC_TO_LDS_B128, double-buffered, drained with s_wait_asynccnt.
// A rows may be padded (reads up to 64-row granule); stores/residual guarded by r<M.
// ACT: 0 none, 1 exact GELU. HASRES: add f32 residual of same shape.
template<int ACT, bool HASRES>
__global__ void gemm_bf16_kernel(const bf16_t* __restrict__ A,
                                 const bf16_t* __restrict__ W,
                                 const float*  __restrict__ bias,
                                 const float*  __restrict__ resid,
                                 float*        __restrict__ out,
                                 int M, int Nc, int K) {
    __shared__ bf16_t smem[2][(MT + NT) * LDS_STRIDE];   // 2 x 15360B

    const int tid  = threadIdx.x;
    const int wave = tid >> 5;
    const int lane = tid & 31;
    const int lo   = lane & 15;
    const int hi   = lane >> 4;
    const int wm   = wave >> 2;       // 0..1
    const int wn   = wave & 3;        // 0..3

    const int nTilesN = Nc / NT;
    const int m0 = (blockIdx.x / nTilesN) * MT;
    const int n0 = (blockIdx.x % nTilesN) * NT;

    // staging assignments (per k-slab)
    const int ar = tid >> 2, ac = tid & 3;           // A: 64 rows x 4 chunks, 1/thread
    const bf16_t* Ag  = A + (size_t)(m0 + ar) * K + ac * 8;
    const bf16_t* Wg0 = W + (size_t)(n0 + ar)      * K + ac * 8;   // B rows 0..63
    const bf16_t* Wg1 = W + (size_t)(n0 + 64 + ar) * K + ac * 8;   // B rows 64..127

    auto stage = [&](int buf, int k0) {
        bf16_t* la = &smem[buf][0];
        bf16_t* lb = &smem[buf][MT * LDS_STRIDE];
        copy_b128_to_lds(Ag  + k0, la + ar * LDS_STRIDE + ac * 8);
        copy_b128_to_lds(Wg0 + k0, lb + ar * LDS_STRIDE + ac * 8);
        copy_b128_to_lds(Wg1 + k0, lb + (64 + ar) * LDS_STRIDE + ac * 8);
    };

    v8f c[2][2] = {};
    const int nk = K / KT;
    int buf = 0;

    stage(0, 0);
    wait_async_copies();
    __syncthreads();

    for (int ks = 0; ks < nk; ++ks) {
        if (ks + 1 < nk) stage(buf ^ 1, (ks + 1) * KT);

        const bf16_t* la = &smem[buf][0];
        const bf16_t* lb = &smem[buf][MT * LDS_STRIDE];

        // A fragments: rows wm*32 + im*16 + lo ; elems (8*hi + e) and (16 + 8*hi + e)
        v16bf af[2];
        #pragma unroll
        for (int im = 0; im < 2; ++im) {
            const bf16_t* p = la + (wm * 32 + im * 16 + lo) * LDS_STRIDE;
            v8bf x0 = *(const v8bf*)(p + 8 * hi);
            v8bf x1 = *(const v8bf*)(p + 16 + 8 * hi);
            #pragma unroll
            for (int e = 0; e < 8; ++e) { af[im][e] = x0[e]; af[im][8 + e] = x1[e]; }
        }
        // B fragments: column = W-row wn*32 + in*16 + lo ; elems (8*hi + e) and (16 + 8*hi + e)
        v16bf bfr[2];
        #pragma unroll
        for (int in = 0; in < 2; ++in) {
            const bf16_t* p = lb + (wn * 32 + in * 16 + lo) * LDS_STRIDE;
            v8bf x0 = *(const v8bf*)(p + 8 * hi);
            v8bf x1 = *(const v8bf*)(p + 16 + 8 * hi);
            #pragma unroll
            for (int e = 0; e < 8; ++e) { bfr[in][e] = x0[e]; bfr[in][8 + e] = x1[e]; }
        }

        #pragma unroll
        for (int im = 0; im < 2; ++im)
            #pragma unroll
            for (int in = 0; in < 2; ++in)
                c[im][in] = __builtin_amdgcn_wmma_f32_16x16x32_bf16(
                    false, af[im], false, bfr[in], (short)0, c[im][in], false, false);

        if (ks + 1 < nk) wait_async_copies();
        __syncthreads();
        buf ^= 1;
    }

    // epilogue: C/D layout -> row = base + v + 8*hi, col = base + lo
    #pragma unroll
    for (int im = 0; im < 2; ++im) {
        #pragma unroll
        for (int in = 0; in < 2; ++in) {
            const int cg = n0 + wn * 32 + in * 16 + lo;
            const float bv = bias ? bias[cg] : 0.f;
            #pragma unroll
            for (int v = 0; v < 8; ++v) {
                const int r = m0 + wm * 32 + im * 16 + v + 8 * hi;
                if (r < M) {
                    float val = c[im][in][v] + bv;
                    if (ACT == 1) val = gelu_exact(val);
                    if (HASRES)   val += resid[(size_t)r * Nc + cg];
                    out[(size_t)r * Nc + cg] = val;
                }
            }
        }
    }
}

// ---------------- divided attention ----------------
// qkv: [B*NTOK, 2304] (q | k | v, head h at column h*64).
// MODE 0: time  (wave per (b,h,sp,fi); L = FC+1 keys: cls + same-sp frames)
// MODE 1: space (wave per (b,h,fi,sp); L = NSPC+1 keys: cls + same-frame spatials)
// MODE 2: cls   (wave per (b,h);      L = NTOK keys: everything)
template<int MODE>
__global__ void attn_kernel(const float* __restrict__ qkv,
                            float* __restrict__ out, int Lpad) {
    extern __shared__ float smem[];
    const int wv   = threadIdx.x >> 5;
    const int lane = threadIdx.x & 31;
    const long wg  = (long)blockIdx.x * 8 + wv;

    int b, h, sp = 0, fi = 0, qtok, L;
    if (MODE == 0) {
        L = FC + 1;
        fi = (int)(wg % FC);  long t = wg / FC;
        sp = (int)(t % NSPC); t /= NSPC;
        h  = (int)(t % HEADS_C); b = (int)(t / HEADS_C);
        qtok = 1 + fi * NSPC + sp;
    } else if (MODE == 1) {
        L = NSPC + 1;
        sp = (int)(wg % NSPC); long t = wg / NSPC;
        fi = (int)(t % FC);    t /= FC;
        h  = (int)(t % HEADS_C); b = (int)(t / HEADS_C);
        qtok = 1 + fi * NSPC + sp;
    } else {
        L = NTOK;
        h = (int)(wg % HEADS_C); b = (int)(wg / HEADS_C);
        qtok = 0;
    }

    auto keytok = [&](int j) -> int {
        if (MODE == 0) return (j == 0) ? 0 : 1 + (j - 1) * NSPC + sp;
        if (MODE == 1) return (j == 0) ? 0 : 1 + fi * NSPC + (j - 1);
        return j;
    };

    float* qs = smem + (size_t)wv * (64 + Lpad);
    float* sc = qs + 64;

    const float scale = 0.125f;  // 64^-0.5
    const float* qp = qkv + (size_t)(b * NTOK + qtok) * QKVC + h * HD;
    qs[lane]      = qp[lane]      * scale;
    qs[lane + 32] = qp[lane + 32] * scale;
    __syncthreads();

    float mx = -3.0e38f;
    for (int j = lane; j < L; j += 32) {
        const int kt = keytok(j);
        const float* kp = qkv + (size_t)(b * NTOK + kt) * QKVC + DIMC + h * HD;
        float s = 0.f;
        #pragma unroll 8
        for (int dd = 0; dd < HD; ++dd) s += qs[dd] * kp[dd];
        sc[j] = s;
        mx = fmaxf(mx, s);
    }
    #pragma unroll
    for (int o = 16; o > 0; o >>= 1) mx = fmaxf(mx, __shfl_xor(mx, o, 32));

    float sum = 0.f;
    for (int j = lane; j < L; j += 32) {
        float e = __expf(sc[j] - mx);
        sc[j] = e;
        sum += e;
    }
    #pragma unroll
    for (int o = 16; o > 0; o >>= 1) sum += __shfl_xor(sum, o, 32);
    const float inv = 1.f / sum;
    __syncthreads();

    float a0 = 0.f, a1 = 0.f;
    for (int j = 0; j < L; ++j) {
        const int kt = keytok(j);
        const float* vp = qkv + (size_t)(b * NTOK + kt) * QKVC + 2 * DIMC + h * HD;
        const float p = sc[j] * inv;
        a0 += p * vp[lane];
        a1 += p * vp[lane + 32];
    }
    float* op = out + (size_t)(b * NTOK + qtok) * DIMC + h * HD;
    op[lane]      = a0;
    op[lane + 32] = a1;
}

// ---------------- host orchestration ----------------
extern "C" void kernel_launch(void* const* d_in, const int* in_sizes, int n_in,
                              void* d_out, int out_size, void* d_ws, size_t ws_size,
                              hipStream_t stream) {
    (void)in_sizes; (void)n_in; (void)out_size; (void)ws_size;

    const float* x       = (const float*)d_in[0];
    const float* Wqkv_t  = (const float*)d_in[1];
    const float* bqkv_t  = (const float*)d_in[2];
    const float* Wproj_t = (const float*)d_in[3];
    const float* bproj_t = (const float*)d_in[4];
    const float* Wqkv_s  = (const float*)d_in[5];
    const float* bqkv_s  = (const float*)d_in[6];
    const float* Wproj_s = (const float*)d_in[7];
    const float* bproj_s = (const float*)d_in[8];
    const float* g1  = (const float*)d_in[9];
    const float* be1 = (const float*)d_in[10];
    const float* g2  = (const float*)d_in[11];
    const float* be2 = (const float*)d_in[12];
    const float* g3  = (const float*)d_in[13];
    const float* be3 = (const float*)d_in[14];
    const float* W1  = (const float*)d_in[15];
    const float* b1  = (const float*)d_in[16];
    const float* W2  = (const float*)d_in[17];
    const float* b2  = (const float*)d_in[18];
    // d_in[19] = seq_len (196), d_in[20] = num_frames (8): fixed at compile time.
    float* out = (float*)d_out;

    const size_t MN  = (size_t)BC * NTOK;            // 25104 rows
    const size_t MNP = ((MN + MT - 1) / MT) * MT;    // 25152, padded for 64-row staging
    char* ws = (char*)d_ws;
    auto al = [](size_t v) { return (v + 255) & ~(size_t)255; };
    size_t off = 0;
    bf16_t* actbf = (bf16_t*)(ws + off); off = al(off + MNP * HIDDENC * sizeof(bf16_t)); // GEMM A (padded)
    float*  bigf  = (float*) (ws + off); off = al(off + MN  * HIDDENC * sizeof(float));  // qkv or h f32
    float*  attn  = (float*) (ws + off); off = al(off + MN * DIMC * sizeof(float));
    float*  tr    = (float*) (ws + off); off = al(off + MN * DIMC * sizeof(float));
    float*  sr    = (float*) (ws + off); off = al(off + MN * DIMC * sizeof(float));
    bf16_t* wQt = (bf16_t*)(ws + off); off = al(off + (size_t)QKVC * DIMC * sizeof(bf16_t));
    bf16_t* wPt = (bf16_t*)(ws + off); off = al(off + (size_t)DIMC * DIMC * sizeof(bf16_t));
    bf16_t* wQs = (bf16_t*)(ws + off); off = al(off + (size_t)QKVC * DIMC * sizeof(bf16_t));
    bf16_t* wPs = (bf16_t*)(ws + off); off = al(off + (size_t)DIMC * DIMC * sizeof(bf16_t));
    bf16_t* w1b = (bf16_t*)(ws + off); off = al(off + (size_t)HIDDENC * DIMC * sizeof(bf16_t));
    bf16_t* w2b = (bf16_t*)(ws + off); off = al(off + (size_t)DIMC * HIDDENC * sizeof(bf16_t));

    const int TPB = 256;
    const int CVG = 2048;

    // weight conversions
    cvt_f32_bf16_kernel<<<CVG, TPB, 0, stream>>>(Wqkv_t,  wQt, (size_t)QKVC * DIMC);
    cvt_f32_bf16_kernel<<<CVG, TPB, 0, stream>>>(Wproj_t, wPt, (size_t)DIMC * DIMC);
    cvt_f32_bf16_kernel<<<CVG, TPB, 0, stream>>>(Wqkv_s,  wQs, (size_t)QKVC * DIMC);
    cvt_f32_bf16_kernel<<<CVG, TPB, 0, stream>>>(Wproj_s, wPs, (size_t)DIMC * DIMC);
    cvt_f32_bf16_kernel<<<CVG, TPB, 0, stream>>>(W1, w1b, (size_t)HIDDENC * DIMC);
    cvt_f32_bf16_kernel<<<CVG, TPB, 0, stream>>>(W2, w2b, (size_t)DIMC * HIDDENC);

    const int mBlocks = (int)(MNP / MT);                 // 393
    const int gQKV  = mBlocks * (QKVC    / NT);          // 393*18
    const int gPROJ = mBlocks * (DIMC    / NT);          // 393*6
    const int gFF1  = mBlocks * (HIDDENC / NT);          // 393*24
    const int gATT  = BC * HEADS_C * NSPC * FC / 8;      // 37632 blocks
    const int gCLS  = BC * HEADS_C / 8;                  // 24 blocks
    const int padT = 32, padS = 224, padC = 1600;
    const size_t smT = 8 * (64 + padT) * sizeof(float);
    const size_t smS = 8 * (64 + padS) * sizeof(float);
    const size_t smC = 8 * (64 + padC) * sizeof(float);  // 53 KB

    // ---- stage 1: time attention ----
    layernorm_bf16_kernel<<<(int)MN, TPB, 0, stream>>>(x, g3, be3, actbf);
    gemm_bf16_kernel<0, false><<<gQKV, TPB, 0, stream>>>(actbf, wQt, bqkv_t, nullptr,
                                                         bigf, (int)MN, QKVC, DIMC);
    attn_kernel<0><<<gATT, TPB, smT, stream>>>(bigf, attn, padT);
    attn_kernel<2><<<gCLS, TPB, smC, stream>>>(bigf, attn, padC);
    cvt_f32_bf16_kernel<<<CVG, TPB, 0, stream>>>(attn, actbf, MN * DIMC);
    gemm_bf16_kernel<0, true><<<gPROJ, TPB, 0, stream>>>(actbf, wPt, bproj_t, x,
                                                         tr, (int)MN, DIMC, DIMC);

    // ---- stage 2: space attention ----
    layernorm_bf16_kernel<<<(int)MN, TPB, 0, stream>>>(tr, g1, be1, actbf);
    gemm_bf16_kernel<0, false><<<gQKV, TPB, 0, stream>>>(actbf, wQs, bqkv_s, nullptr,
                                                         bigf, (int)MN, QKVC, DIMC);
    attn_kernel<1><<<gATT, TPB, smS, stream>>>(bigf, attn, padS);
    attn_kernel<2><<<gCLS, TPB, smC, stream>>>(bigf, attn, padC);
    cvt_f32_bf16_kernel<<<CVG, TPB, 0, stream>>>(attn, actbf, MN * DIMC);
    gemm_bf16_kernel<0, true><<<gPROJ, TPB, 0, stream>>>(actbf, wPs, bproj_s, tr,
                                                         sr, (int)MN, DIMC, DIMC);

    // ---- MLP ----
    layernorm_bf16_kernel<<<(int)MN, TPB, 0, stream>>>(sr, g2, be2, actbf);
    gemm_bf16_kernel<1, false><<<gFF1, TPB, 0, stream>>>(actbf, w1b, b1, nullptr,
                                                         bigf, (int)MN, HIDDENC, DIMC);
    cvt_f32_bf16_kernel<<<CVG, TPB, 0, stream>>>(bigf, actbf, MN * HIDDENC);
    gemm_bf16_kernel<0, true><<<gPROJ, TPB, 0, stream>>>(actbf, w2b, b2, sr,
                                                         out, (int)MN, DIMC, HIDDENC);
}
